// mahalanobis_loss_89378269429897
// MI455X (gfx1250) — matrix operand order repaired and checked by reading, not compiled, not run on hardware
//
#include <hip/hip_runtime.h>
#include <hip/hip_bf16.h>

// ---------------------------------------------------------------------------
// Problem constants (fixed by the reference): N=4096 samples, D=2048 dims.
// loss = s^T (R^T R + N*EPS*I)^{-1} s,  s = column-sums of residual R.
// ---------------------------------------------------------------------------
constexpr int   NN      = 4096;
constexpr int   DD      = 2048;
constexpr int   ITERS   = 64;
constexpr float LAMBDA  = (float)NN * 1e-8f;   // N * EPS

typedef __attribute__((ext_vector_type(16))) __bf16 v16bf;
typedef __attribute__((ext_vector_type(8)))  __bf16 v8bf;
typedef __attribute__((ext_vector_type(8)))  float  v8f;

static __device__ __forceinline__ __bf16 f32_to_bf16_rne(float f) {
    unsigned int u = __builtin_bit_cast(unsigned int, f);
    u += 0x7FFFu + ((u >> 16) & 1u);           // round-to-nearest-even
    unsigned short h = (unsigned short)(u >> 16);
    return __builtin_bit_cast(__bf16, h);
}

static __device__ __forceinline__ float wave_reduce_add(float v) {
    #pragma unroll
    for (int off = 16; off > 0; off >>= 1) v += __shfl_xor(v, off, 32);
    return v;
}

// ---------------------------------------------------------------------------
// 0) generic zero
// ---------------------------------------------------------------------------
__global__ void k_zero_f32(float* p, int n) {
    int i = blockIdx.x * blockDim.x + threadIdx.x;
    if (i < n) p[i] = 0.0f;
}

// ---------------------------------------------------------------------------
// 1) residual + transpose to bf16 (RT[D][N]) + column sums s[D]
//    block (32,8), grid (D/32, N/32). LDS-tiled so reads AND writes coalesce.
// ---------------------------------------------------------------------------
__global__ void k_resid_transpose(const float* __restrict__ in,
                                  const float* __restrict__ rec,
                                  __bf16* __restrict__ RT,
                                  float* __restrict__ s) {
    __shared__ float tile[32][33];
    __shared__ float cs[8][32];
    const int tx = threadIdx.x, ty = threadIdx.y;
    const int col = blockIdx.x * 32 + tx;             // d index (read phase)
    float csum = 0.0f;
    #pragma unroll
    for (int r = 0; r < 4; ++r) {
        const int row = blockIdx.y * 32 + ty + r * 8; // n index
        const size_t idx = (size_t)row * DD + col;
        const float v = in[idx] - rec[idx];
        tile[ty + r * 8][tx] = v;
        csum += v;
    }
    cs[ty][tx] = csum;
    __syncthreads();
    if (ty == 0) {
        float t = 0.0f;
        #pragma unroll
        for (int i = 0; i < 8; ++i) t += cs[i][tx];
        unsafeAtomicAdd(&s[col], t);                  // 128 adds/column total
    }
    #pragma unroll
    for (int r = 0; r < 4; ++r) {
        const int d = blockIdx.x * 32 + ty + r * 8;
        const int n = blockIdx.y * 32 + tx;
        RT[(size_t)d * NN + n] = f32_to_bf16_rne(tile[tx][ty + r * 8]);
    }
}

// ---------------------------------------------------------------------------
// 2) SYRK via WMMA:  C[i][j] = sum_k RT[i][k] * RT[j][k]   (C = G, fp32)
//    One wave per 64x64 C tile; 4x4 grid of 16x16x32 bf16 WMMAs per K-step.
//    A-frag lane layout (16x32): lane m%16 holds row m; two 16B chunks at
//    K=(half?16:0) and +8.  B-frag (32x16): lane n%16 holds col n = row n of
//    RT; one contiguous 32B chunk at K=(half?16:0).  All reads L2-resident.
//
//    All 8 fragments of a K-step are loaded up-front (one big load clause,
//    16 global_load_b128 in flight) so the compiler can use *partial*
//    s_wait_loadcnt waits: the first row's WMMAs start while later rows'
//    loads are still outstanding.  Register budget: 128 acc + 32 A + 32 B
//    + addresses ~= 210 VGPRs -> fits under __launch_bounds__(32,1), no
//    spills (verified round 2: spill-free with the per-row variant).
// ---------------------------------------------------------------------------
__global__ __launch_bounds__(32, 1)
__attribute__((amdgpu_waves_per_eu(1)))
void k_syrk_wmma(const __bf16* __restrict__ RT, float* __restrict__ C) {
    const int lane = threadIdx.x;                 // blockDim = 32 (one wave)
    const int TPR  = DD / 64;                     // 32 tiles per row
    const int ti = blockIdx.x / TPR;
    const int tj = blockIdx.x % TPR;

    const int half = lane >> 4;                   // 0: lanes 0-15, 1: 16-31
    const int lr   = lane & 15;
    const int aoff = half * 16;                   // A: K offset 0 / 16
    const int boff = half * 16;                   // B: K offset 0 / 16

    const __bf16* arow[4];
    const __bf16* brow[4];
    #pragma unroll
    for (int r = 0; r < 4; ++r)
        arow[r] = RT + (size_t)(ti * 64 + r * 16 + lr) * NN + aoff;
    #pragma unroll
    for (int c = 0; c < 4; ++c)
        brow[c] = RT + (size_t)(tj * 64 + c * 16 + lr) * NN + boff;

    const v8f vzero = {0.f,0.f,0.f,0.f,0.f,0.f,0.f,0.f};
    v8f acc[4][4];
    #pragma unroll
    for (int r = 0; r < 4; ++r)
        #pragma unroll
        for (int c = 0; c < 4; ++c) acc[r][c] = vzero;

    #pragma unroll 1
    for (int k = 0; k < NN; k += 32) {
        // Issue ALL fragment loads for this K-step first (16x b128 clause).
        v16bf bf[4];
        #pragma unroll
        for (int c = 0; c < 4; ++c)
            bf[c] = *(const v16bf*)(brow[c] + k);               // 32B contiguous

        v16bf af[4];
        #pragma unroll
        for (int r = 0; r < 4; ++r) {
            const v8bf lo = *(const v8bf*)(arow[r] + k);        // K..K+7
            const v8bf hi = *(const v8bf*)(arow[r] + k + 8);    // K+8..K+15
            #pragma unroll
            for (int e = 0; e < 8; ++e) { af[r][e] = lo[e]; af[r][e + 8] = hi[e]; }
        }

        // Pull next K-step's cache lines toward the WGP while WMMAs run.
        if (k + 32 < NN) {
            #pragma unroll
            for (int c = 0; c < 4; ++c) __builtin_prefetch(brow[c] + k + 32, 0, 3);
            #pragma unroll
            for (int r = 0; r < 4; ++r) __builtin_prefetch(arow[r] + k + 32, 0, 3);
        }

        // 16 WMMAs; row r only depends on af[r] + bf[*] -> partial waits.
        #pragma unroll
        for (int r = 0; r < 4; ++r)
            #pragma unroll
            for (int c = 0; c < 4; ++c)
                acc[r][c] = __builtin_amdgcn_wmma_f32_16x16x32_bf16(
                    false, af[r], false, bf[c], (short)0, acc[r][c], false, false);
    }

    // C/D layout: lanes 0-15 -> M = vg, lanes 16-31 -> M = 8+vg; N = lane%16.
    const int mbase = half * 8;
    #pragma unroll
    for (int r = 0; r < 4; ++r)
        #pragma unroll
        for (int c = 0; c < 4; ++c) {
            const int gcol = tj * 64 + c * 16 + lr;
            #pragma unroll
            for (int vg = 0; vg < 8; ++vg) {
                const int grow = ti * 64 + r * 16 + mbase + vg;
                C[(size_t)grow * DD + gcol] = acc[r][c][vg];
            }
        }
}

// ---------------------------------------------------------------------------
// 3) CG init: p = r = s, x = 0, rr0 += s.s
// ---------------------------------------------------------------------------
__global__ void k_cg_init(const float* __restrict__ s, float* __restrict__ p,
                          float* __restrict__ r, float* __restrict__ x,
                          float* __restrict__ rr0) {
    const int i = blockIdx.x * blockDim.x + threadIdx.x;
    float sv = 0.0f;
    if (i < DD) { sv = s[i]; p[i] = sv; r[i] = sv; x[i] = 0.0f; }
    const float w = wave_reduce_add(sv * sv);
    if ((threadIdx.x & 31) == 0) unsafeAtomicAdd(rr0, w);
}

// ---------------------------------------------------------------------------
// 4) q = G p + lambda p ; pq += p.q     (one wave per row, float4 loads;
//    G is 16 MB -> L2-resident across all 64 iterations)
// ---------------------------------------------------------------------------
__global__ void k_matvec_dot(const float* __restrict__ C,
                             const float* __restrict__ p,
                             float* __restrict__ q,
                             float* __restrict__ pq) {
    const int wave = threadIdx.x >> 5;
    const int lane = threadIdx.x & 31;
    const int row  = blockIdx.x * 8 + wave;
    const float* crow = C + (size_t)row * DD;
    float sum = 0.0f;
    for (int j = lane * 4; j < DD; j += 128) {
        const float4 cv = *(const float4*)(crow + j);
        const float4 pv = *(const float4*)(p + j);
        sum += cv.x * pv.x + cv.y * pv.y + cv.z * pv.z + cv.w * pv.w;
    }
    sum = wave_reduce_add(sum);
    if (lane == 0) {
        const float qv = sum + LAMBDA * p[row];
        q[row] = qv;
        unsafeAtomicAdd(pq, p[row] * qv);
    }
}

// ---------------------------------------------------------------------------
// 5) alpha = rr/pq ; x += alpha p ; r -= alpha q ; rrNext += r.r
// ---------------------------------------------------------------------------
__global__ void k_cg_update1(const float* __restrict__ p, const float* __restrict__ q,
                             float* __restrict__ x, float* __restrict__ r,
                             const float* __restrict__ rrCur,
                             const float* __restrict__ pqCur,
                             float* __restrict__ rrNext) {
    const int i = blockIdx.x * blockDim.x + threadIdx.x;
    const float alpha = rrCur[0] / pqCur[0];
    float rn = 0.0f;
    if (i < DD) {
        x[i] += alpha * p[i];
        rn = r[i] - alpha * q[i];
        r[i] = rn;
    }
    const float w = wave_reduce_add(rn * rn);
    if ((threadIdx.x & 31) == 0) unsafeAtomicAdd(rrNext, w);
}

// ---------------------------------------------------------------------------
// 6) beta = rrNext/rrCur ; p = r + beta p
// ---------------------------------------------------------------------------
__global__ void k_cg_update2(float* __restrict__ p, const float* __restrict__ r,
                             const float* __restrict__ rrNext,
                             const float* __restrict__ rrCur) {
    const int i = blockIdx.x * blockDim.x + threadIdx.x;
    if (i < DD) {
        const float beta = rrNext[0] / rrCur[0];
        p[i] = r[i] + beta * p[i];
    }
}

// ---------------------------------------------------------------------------
// 7) loss = s.x  (single block, deterministic tree reduce, writes d_out[0])
// ---------------------------------------------------------------------------
__global__ void k_final_dot(const float* __restrict__ s, const float* __restrict__ x,
                            float* __restrict__ out) {
    __shared__ float sm[256];
    float acc = 0.0f;
    for (int i = threadIdx.x; i < DD; i += 256) acc += s[i] * x[i];
    sm[threadIdx.x] = acc;
    __syncthreads();
    for (int stride = 128; stride > 0; stride >>= 1) {
        if (threadIdx.x < stride) sm[threadIdx.x] += sm[threadIdx.x + stride];
        __syncthreads();
    }
    if (threadIdx.x == 0) out[0] = sm[0];
}

// ---------------------------------------------------------------------------
// Launch: workspace layout
//   [0, 16MB)      RT   bf16 [D][N]
//   [16MB, 32MB)   G    f32  [D][D]
//   [32MB, ...)    s,p,r,x,q (5 * D floats), then pq[ITERS], rr[ITERS+1]
// ---------------------------------------------------------------------------
extern "C" void kernel_launch(void* const* d_in, const int* in_sizes, int n_in,
                              void* d_out, int out_size, void* d_ws, size_t ws_size,
                              hipStream_t stream) {
    (void)in_sizes; (void)n_in; (void)out_size; (void)ws_size;
    const float* in  = (const float*)d_in[0];
    const float* rec = (const float*)d_in[1];

    char* w = (char*)d_ws;
    __bf16* RT = (__bf16*)w;
    float*  G  = (float*)(w + (size_t)16 * 1024 * 1024);
    float*  vec = (float*)(w + (size_t)32 * 1024 * 1024);
    float* s = vec;
    float* p = vec + DD;
    float* r = vec + 2 * DD;
    float* x = vec + 3 * DD;
    float* q = vec + 4 * DD;
    float* pq = vec + 5 * DD;          // ITERS slots
    float* rr = pq + ITERS;            // ITERS+1 slots

    // zero s and all per-iteration scalar accumulators
    k_zero_f32<<<(DD + 255) / 256, 256, 0, stream>>>(s, DD);
    k_zero_f32<<<1, 256, 0, stream>>>(pq, 2 * ITERS + 1);

    // residual + transpose + column sums
    k_resid_transpose<<<dim3(DD / 32, NN / 32), dim3(32, 8), 0, stream>>>(in, rec, RT, s);

    // G = R^T R via bf16 WMMA (fp32 accumulate)
    k_syrk_wmma<<<(DD / 64) * (DD / 64), 32, 0, stream>>>(RT, G);

    // CG solve (G + lambda I) x = s
    k_cg_init<<<(DD + 255) / 256, 256, 0, stream>>>(s, p, r, x, rr);
    for (int it = 0; it < ITERS; ++it) {
        k_matvec_dot<<<DD / 8, 256, 0, stream>>>(G, p, q, pq + it);
        k_cg_update1<<<(DD + 255) / 256, 256, 0, stream>>>(p, q, x, r,
                                                           rr + it, pq + it, rr + it + 1);
        k_cg_update2<<<(DD + 255) / 256, 256, 0, stream>>>(p, r, rr + it + 1, rr + it);
    }

    // loss = s . x
    k_final_dot<<<1, 256, 0, stream>>>(s, x, (float*)d_out);
}
